// RecurrentAutoencoder_12807592476818
// MI455X (gfx1250) — matrix operand order repaired and checked by reading, not compiled, not run on hardware
//
#include <hip/hip_runtime.h>

typedef __bf16 bf16;
typedef __attribute__((ext_vector_type(8)))  __bf16 v8bf;
typedef __attribute__((ext_vector_type(16))) __bf16 v16bf;
typedef __attribute__((ext_vector_type(8)))  float  v8f;

#define B_   64
#define T_   256
#define D_   64
#define H_   512
#define Z_   128
#define G_   1536   // 3*H

__device__ __forceinline__ v8f zero_v8f() {
  v8f v;
#pragma unroll
  for (int i = 0; i < 8; ++i) v[i] = 0.f;
  return v;
}
// Add an opaque zero offset: keeps the pointer's provenance (so the loads
// stay global_load, not flat_load) while making the address depend on a
// per-iteration volatile asm result, so weight loads cannot be LICM-hoisted
// out of the sequential time loop and spilled to scratch.
__device__ __forceinline__ const bf16* opaque(const bf16* p) {
  size_t z = 0;
  asm volatile("" : "+s"(z));
  return p + z;
}
// 16-bit A fragment 16x32: lane m=lane&15, elements 0..7 = K[ko..ko+7],
// 8..15 = K[ko+16..ko+23], ko = (lane>>4)*8  (ISA 7.12.2)
__device__ __forceinline__ v16bf frag_a(const bf16* p) {
  v8bf lo = *(const v8bf*)(p);
  v8bf hi = *(const v8bf*)(p + 16);
  return __builtin_shufflevector(lo, hi, 0,1,2,3,4,5,6,7,8,9,10,11,12,13,14,15);
}
// 16-bit B fragment 32x16: lane col n=lane&15, elements 0..7 = K[kb..kb+7],
// 8..15 = K[kb+16..kb+23], kb = (lane>>4)*8 (same order as A)
__device__ __forceinline__ v16bf frag_b(const bf16* p) {
  v8bf lo = *(const v8bf*)(p);
  v8bf hi = *(const v8bf*)(p + 16);
  return __builtin_shufflevector(lo, hi, 0,1,2,3,4,5,6,7,8,9,10,11,12,13,14,15);
}
__device__ __forceinline__ v8f wmma_bf(v16bf a, v16bf b, v8f c) {
  return __builtin_amdgcn_wmma_f32_16x16x32_bf16(false, a, false, b, (short)0, c,
                                                 false, false);
}
__device__ __forceinline__ float sigmf(float x) { return 1.f / (1.f + __expf(-x)); }

// ---------------------------------------------------------------- converts
__global__ __launch_bounds__(256) void cvt_bf16(const float* __restrict__ s,
                                                bf16* __restrict__ d, int n) {
  int i = blockIdx.x * 256 + threadIdx.x;
  if (i < n) d[i] = (bf16)s[i];
}
// x (B,T,D) f32 -> (T,B,D) bf16
__global__ __launch_bounds__(256) void x_to_bf16_tbd(const float* __restrict__ x,
                                                     bf16* __restrict__ xbf) {
  int i = blockIdx.x * 256 + threadIdx.x;
  if (i >= T_ * B_ * D_) return;
  int d = i & 63;
  int b = (i >> 6) & 63;
  int t = i >> 12;
  xbf[i] = (bf16)x[((size_t)b * T_ + t) * D_ + d];
}

// ---------------------------------------------------------------- tiled GEMM
// C[M,N] = A[M,K] @ W[N,K]^T + bias ; A,W bf16 row-major, one 16x16 tile/wave
template <bool OUT_BF16>
__global__ __launch_bounds__(256) void gemm_wmma(
    const bf16* __restrict__ A, int lda, const bf16* __restrict__ W,
    const float* __restrict__ bias, void* __restrict__ Cv, int ldc,
    int M, int N, int K) {
  const int wave = threadIdx.x >> 5, lane = threadIdx.x & 31;
  const int nt_n = N >> 4;
  const int tile = blockIdx.x * 8 + wave;
  if (tile >= (M >> 4) * nt_n) return;
  const int mt = tile / nt_n, nt = tile - mt * nt_n;
  const int nl = lane & 15;
  const bf16* a_ptr = A + (size_t)(mt * 16 + nl) * lda + ((lane >> 4) << 3);
  const bf16* b_ptr = W + (size_t)(nt * 16 + nl) * K + ((lane >> 4) << 3);
  v8f acc = zero_v8f();
  for (int k = 0; k < K; k += 32)
    acc = wmma_bf(frag_a(a_ptr + k), frag_b(b_ptr + k), acc);
  const int n = nt * 16 + nl;
  const float bv = bias ? bias[n] : 0.f;
  const int m0 = mt * 16 + ((lane >> 4) << 3);
#pragma unroll
  for (int r = 0; r < 8; ++r) {
    float v = acc[r] + bv;
    if (OUT_BF16) ((bf16*)Cv)[(size_t)(m0 + r) * ldc + n] = (bf16)v;
    else          ((float*)Cv)[(size_t)(m0 + r) * ldc + n] = v;
  }
}

// ---------------------------------------------------------------- encoder scan
// One WG = one 16-row batch tile of one direction. 8 waves x 4 tiles cover 3H
// gate columns grouped as (r,z,n) triples sharing C-fragment positions.
__global__ __launch_bounds__(256) void enc_scan(
    const bf16* __restrict__ gx_f, const bf16* __restrict__ gx_b,
    const bf16* __restrict__ Whh_f, const bf16* __restrict__ Whh_b,
    const float* __restrict__ bhh_f, const float* __restrict__ bhh_b,
    bf16* __restrict__ ys,   // (T,B,2H) bf16
    float* __restrict__ hT)  // (B,2H) f32
{
  const int dir = blockIdx.x & 1;
  const int bt = blockIdx.x >> 1;
  const bf16*  __restrict__ gx  = dir ? gx_b : gx_f;
  const bf16*  __restrict__ Whh = dir ? Whh_b : Whh_f;
  const float* __restrict__ bhh = dir ? bhh_b : bhh_f;
  const int wave = threadIdx.x >> 5, lane = threadIdx.x & 31;
  const int nl = lane & 15;
  const int mb = (lane >> 4) << 3;
  const int koA = (lane >> 4) << 3;
  const int koB = (lane >> 4) << 3;
  __shared__ bf16 hbf[16 * 520];
  float h[4][8];
#pragma unroll
  for (int tile = 0; tile < 4; ++tile)
#pragma unroll
    for (int r = 0; r < 8; ++r) h[tile][r] = 0.f;
  for (int i = threadIdx.x; i < 16 * 520; i += 256) hbf[i] = (bf16)0.f;
  __syncthreads();

  for (int s = 0; s < T_; ++s) {
    const int ti = dir ? (T_ - 1 - s) : s;
    const bf16* Whh_t = opaque(Whh);        // defeat cross-step hoist/spill
    const bf16* gxt = opaque(gx + ((size_t)ti * B_ + bt * 16) * G_);
    if (s + 1 < T_) {
      const int tn = dir ? (T_ - 2 - s) : (s + 1);
      __builtin_prefetch(
          (const char*)(gx + ((size_t)tn * B_ + bt * 16) * G_) + threadIdx.x * 128,
          0, 0);
    }
    float hn[4][8];
#pragma unroll
    for (int tile = 0; tile < 4; ++tile) {
      const int n = wave * 64 + tile * 16 + nl;
      v8f ar = zero_v8f(), az = zero_v8f(), an = zero_v8f();
      const bf16* ap = hbf + nl * 520 + koA;
      const bf16* bp = Whh_t + (size_t)n * 512 + koB;
      for (int k = 0; k < 512; k += 32) {
        v16bf a = frag_a(ap + k);
        ar = wmma_bf(a, frag_b(bp + k), ar);
        az = wmma_bf(a, frag_b(bp + 512 * 512 + k), az);
        an = wmma_bf(a, frag_b(bp + 1024 * 512 + k), an);
      }
      const float br = bhh[n], bz = bhh[512 + n], bn = bhh[1024 + n];
#pragma unroll
      for (int r = 0; r < 8; ++r) {
        const bf16* gp = gxt + (size_t)(mb + r) * G_ + n;
        const float rr = sigmf((float)gp[0] + ar[r] + br);
        const float zz = sigmf((float)gp[512] + az[r] + bz);
        const float nn = tanhf((float)gp[1024] + rr * (an[r] + bn));
        hn[tile][r] = (1.f - zz) * nn + zz * h[tile][r];
        h[tile][r] = hn[tile][r];
      }
    }
    __syncthreads();  // all reads of old h done
#pragma unroll
    for (int tile = 0; tile < 4; ++tile) {
      const int n = wave * 64 + tile * 16 + nl;
#pragma unroll
      for (int r = 0; r < 8; ++r) {
        const int m = mb + r;
        const bf16 hv = (bf16)hn[tile][r];
        hbf[m * 520 + n] = hv;
        ys[((size_t)ti * B_ + bt * 16 + m) * (2 * H_) + dir * H_ + n] = hv;
      }
    }
    __syncthreads();
  }
#pragma unroll
  for (int tile = 0; tile < 4; ++tile) {
    const int n = wave * 64 + tile * 16 + nl;
#pragma unroll
    for (int r = 0; r < 8; ++r)
      hT[(size_t)(bt * 16 + mb + r) * (2 * H_) + dir * H_ + n] = h[tile][r];
  }
}

// ---------------------------------------------------------------- latent LN
__global__ __launch_bounds__(128) void latent_ln(
    const float* __restrict__ zpre, const float* __restrict__ g,
    const float* __restrict__ bb, float* __restrict__ zout,
    bf16* __restrict__ zbf) {
  const int b = blockIdx.x, j = threadIdx.x;
  __shared__ float red[128];
  const float v = zpre[b * 128 + j];
  red[j] = v;
  __syncthreads();
  for (int s = 64; s > 0; s >>= 1) {
    if (j < s) red[j] += red[j + s];
    __syncthreads();
  }
  const float mu = red[0] * (1.f / 128.f);
  __syncthreads();
  const float dv = v - mu;
  red[j] = dv * dv;
  __syncthreads();
  for (int s = 64; s > 0; s >>= 1) {
    if (j < s) red[j] += red[j + s];
    __syncthreads();
  }
  const float var = red[0] * (1.f / 128.f);
  const float zt = tanhf(dv * rsqrtf(var + 1e-5f) * g[j] + bb[j]);
  zout[b * 128 + j] = zt;
  zbf[b * 128 + j] = (bf16)zt;
}

// ---------------------------------------------------------------- decoder scan
// Autoregressive: input GEMM fused into the step. n-gate keeps gx_n / gh_n
// accumulators separate (n = tanh(gx_n + r*gh_n)).
__global__ __launch_bounds__(256) void dec_scan(
    const float* __restrict__ x,    // (B,T,D) f32 (inp0 = x[:,0,:])
    const float* __restrict__ hid,  // (B, 2H) f32 initial states
    const bf16* __restrict__ Wih1, const bf16* __restrict__ Whh1,
    const float* __restrict__ bih1, const float* __restrict__ bhh1,
    const bf16* __restrict__ Wih2, const bf16* __restrict__ Whh2,
    const float* __restrict__ bih2, const float* __restrict__ bhh2,
    const bf16* __restrict__ Wout, const float* __restrict__ bout,
    float* __restrict__ rec)        // (B,T,D)
{
  const int bt = blockIdx.x;
  const int wave = threadIdx.x >> 5, lane = threadIdx.x & 31;
  const int nl = lane & 15;
  const int mb = (lane >> 4) << 3;
  const int koA = (lane >> 4) << 3;
  const int koB = (lane >> 4) << 3;
  __shared__ bf16 inp[16 * 72];
  __shared__ bf16 h1b[16 * 520];
  __shared__ bf16 h2b[16 * 520];
  float h1[4][8], h2[4][8];
#pragma unroll
  for (int tile = 0; tile < 4; ++tile) {
    const int n = wave * 64 + tile * 16 + nl;
#pragma unroll
    for (int r = 0; r < 8; ++r) {
      const int m = mb + r;
      const size_t b = (size_t)(bt * 16 + m);
      h1[tile][r] = hid[b * 1024 + n];
      h2[tile][r] = hid[b * 1024 + 512 + n];
      h1b[m * 520 + n] = (bf16)h1[tile][r];
      h2b[m * 520 + n] = (bf16)h2[tile][r];
    }
  }
  for (int i = threadIdx.x; i < 16 * 64; i += 256) {
    const int m = i >> 6, d = i & 63;
    inp[m * 72 + d] = (bf16)x[(size_t)(bt * 16 + m) * (T_ * D_) + d];
  }
  __syncthreads();

  for (int t = 0; t < T_; ++t) {
    // stage 1: h1 = GRU1(inp, h1)
    float h1n[4][8];
    {
      const bf16* Wih1_t = opaque(Wih1);
      const bf16* Whh1_t = opaque(Whh1);
#pragma unroll
      for (int tile = 0; tile < 4; ++tile) {
        const int n = wave * 64 + tile * 16 + nl;
        v8f ar = zero_v8f(), az = zero_v8f(), axn = zero_v8f(), ahn = zero_v8f();
        {
          const bf16* ap = inp + nl * 72 + koA;
          const bf16* bp = Wih1_t + (size_t)n * 64 + koB;
#pragma unroll
          for (int k = 0; k < 64; k += 32) {
            v16bf a = frag_a(ap + k);
            ar  = wmma_bf(a, frag_b(bp + k), ar);
            az  = wmma_bf(a, frag_b(bp + 512 * 64 + k), az);
            axn = wmma_bf(a, frag_b(bp + 1024 * 64 + k), axn);
          }
        }
        {
          const bf16* ap = h1b + nl * 520 + koA;
          const bf16* bp = Whh1_t + (size_t)n * 512 + koB;
          for (int k = 0; k < 512; k += 32) {
            v16bf a = frag_a(ap + k);
            ar  = wmma_bf(a, frag_b(bp + k), ar);
            az  = wmma_bf(a, frag_b(bp + 512 * 512 + k), az);
            ahn = wmma_bf(a, frag_b(bp + 1024 * 512 + k), ahn);
          }
        }
        const float br = bih1[n] + bhh1[n];
        const float bz = bih1[512 + n] + bhh1[512 + n];
        const float bin = bih1[1024 + n], bhn = bhh1[1024 + n];
#pragma unroll
        for (int r = 0; r < 8; ++r) {
          const float rr = sigmf(ar[r] + br);
          const float zz = sigmf(az[r] + bz);
          const float nn = tanhf(axn[r] + bin + rr * (ahn[r] + bhn));
          h1n[tile][r] = (1.f - zz) * nn + zz * h1[tile][r];
          h1[tile][r] = h1n[tile][r];
        }
      }
    }
    __syncthreads();
#pragma unroll
    for (int tile = 0; tile < 4; ++tile) {
      const int n = wave * 64 + tile * 16 + nl;
#pragma unroll
      for (int r = 0; r < 8; ++r) h1b[(mb + r) * 520 + n] = (bf16)h1n[tile][r];
    }
    __syncthreads();

    // stage 2: h2 = GRU2(h1, h2)
    float h2n[4][8];
    {
      const bf16* Wih2_t = opaque(Wih2);
      const bf16* Whh2_t = opaque(Whh2);
#pragma unroll
      for (int tile = 0; tile < 4; ++tile) {
        const int n = wave * 64 + tile * 16 + nl;
        v8f ar = zero_v8f(), az = zero_v8f(), axn = zero_v8f(), ahn = zero_v8f();
        {
          const bf16* ap = h1b + nl * 520 + koA;
          const bf16* bp = Wih2_t + (size_t)n * 512 + koB;
          for (int k = 0; k < 512; k += 32) {
            v16bf a = frag_a(ap + k);
            ar  = wmma_bf(a, frag_b(bp + k), ar);
            az  = wmma_bf(a, frag_b(bp + 512 * 512 + k), az);
            axn = wmma_bf(a, frag_b(bp + 1024 * 512 + k), axn);
          }
        }
        {
          const bf16* ap = h2b + nl * 520 + koA;
          const bf16* bp = Whh2_t + (size_t)n * 512 + koB;
          for (int k = 0; k < 512; k += 32) {
            v16bf a = frag_a(ap + k);
            ar  = wmma_bf(a, frag_b(bp + k), ar);
            az  = wmma_bf(a, frag_b(bp + 512 * 512 + k), az);
            ahn = wmma_bf(a, frag_b(bp + 1024 * 512 + k), ahn);
          }
        }
        const float br = bih2[n] + bhh2[n];
        const float bz = bih2[512 + n] + bhh2[512 + n];
        const float bin = bih2[1024 + n], bhn = bhh2[1024 + n];
#pragma unroll
        for (int r = 0; r < 8; ++r) {
          const float rr = sigmf(ar[r] + br);
          const float zz = sigmf(az[r] + bz);
          const float nn = tanhf(axn[r] + bin + rr * (ahn[r] + bhn));
          h2n[tile][r] = (1.f - zz) * nn + zz * h2[tile][r];
          h2[tile][r] = h2n[tile][r];
        }
      }
    }
    __syncthreads();
#pragma unroll
    for (int tile = 0; tile < 4; ++tile) {
      const int n = wave * 64 + tile * 16 + nl;
#pragma unroll
      for (int r = 0; r < 8; ++r) h2b[(mb + r) * 520 + n] = (bf16)h2n[tile][r];
    }
    __syncthreads();

    // stage 3: out = h2 @ Wout^T + bout ; feed back as inp
    if (wave < 4) {
      const bf16* Wout_t = opaque(Wout);
      const int n = wave * 16 + nl;
      v8f ao = zero_v8f();
      const bf16* ap = h2b + nl * 520 + koA;
      const bf16* bp = Wout_t + (size_t)n * 512 + koB;
      for (int k = 0; k < 512; k += 32)
        ao = wmma_bf(frag_a(ap + k), frag_b(bp + k), ao);
      const float bv = bout[n];
#pragma unroll
      for (int r = 0; r < 8; ++r) {
        const float ov = ao[r] + bv;
        rec[((size_t)(bt * 16 + mb + r) * T_ + t) * D_ + n] = ov;
        inp[(mb + r) * 72 + n] = (bf16)ov;
      }
    }
    __syncthreads();
  }
}

// ---------------------------------------------------------------- launcher
extern "C" void kernel_launch(void* const* d_in, const int* in_sizes, int n_in,
                              void* d_out, int out_size, void* d_ws,
                              size_t ws_size, hipStream_t stream) {
  (void)in_sizes; (void)n_in; (void)out_size; (void)ws_size;
  const float* X = (const float*)d_in[0];  // (B,T,D)
  auto ENC = [&](int l, int dir, int w) {
    return (const float*)d_in[2 + (l * 2 + dir) * 4 + w];
  };
  auto DEC = [&](int l, int w) { return (const float*)d_in[26 + l * 4 + w]; };
  const float* Wlat = (const float*)d_in[34];
  const float* blat = (const float*)d_in[35];
  const float* lng  = (const float*)d_in[36];
  const float* lnb  = (const float*)d_in[37];
  const float* Wz2h = (const float*)d_in[38];
  const float* bz2h = (const float*)d_in[39];
  const float* Wout = (const float*)d_in[40];
  const float* bout = (const float*)d_in[41];

  char* p = (char*)d_ws;
  auto alloc = [&](size_t bytes) {
    char* r = p;
    p += (bytes + 255) & ~(size_t)255;
    return r;
  };
  bf16*  xbf    = (bf16*)alloc((size_t)T_ * B_ * D_ * 2);
  bf16*  gx0    = (bf16*)alloc((size_t)T_ * B_ * G_ * 2);
  bf16*  gx1    = (bf16*)alloc((size_t)T_ * B_ * G_ * 2);
  bf16*  ysA    = (bf16*)alloc((size_t)T_ * B_ * 2 * H_ * 2);
  bf16*  ysB    = (bf16*)alloc((size_t)T_ * B_ * 2 * H_ * 2);
  float* comb   = (float*)alloc((size_t)B_ * 2 * H_ * 4);
  bf16*  combbf = (bf16*)alloc((size_t)B_ * 2 * H_ * 2);
  float* zpre   = (float*)alloc((size_t)B_ * Z_ * 4);
  bf16*  zbf    = (bf16*)alloc((size_t)B_ * Z_ * 2);
  float* hid    = (float*)alloc((size_t)B_ * 2 * H_ * 4);
  bf16 *wih[3][2], *whh[3][2];
  for (int l = 0; l < 3; ++l)
    for (int d = 0; d < 2; ++d) {
      wih[l][d] = (bf16*)alloc((size_t)G_ * (l ? 2 * H_ : D_) * 2);
      whh[l][d] = (bf16*)alloc((size_t)G_ * H_ * 2);
    }
  bf16* dwih1 = (bf16*)alloc((size_t)G_ * D_ * 2);
  bf16* dwhh1 = (bf16*)alloc((size_t)G_ * H_ * 2);
  bf16* dwih2 = (bf16*)alloc((size_t)G_ * H_ * 2);
  bf16* dwhh2 = (bf16*)alloc((size_t)G_ * H_ * 2);
  bf16* woutb = (bf16*)alloc((size_t)D_ * H_ * 2);
  bf16* wlatb = (bf16*)alloc((size_t)Z_ * 2 * H_ * 2);
  bf16* wz2hb = (bf16*)alloc((size_t)2 * H_ * Z_ * 2);

  auto conv = [&](const float* s, bf16* d, int n) {
    cvt_bf16<<<(n + 255) / 256, 256, 0, stream>>>(s, d, n);
  };
  x_to_bf16_tbd<<<(T_ * B_ * D_) / 256, 256, 0, stream>>>(X, xbf);
  for (int l = 0; l < 3; ++l)
    for (int d = 0; d < 2; ++d) {
      conv(ENC(l, d, 0), wih[l][d], G_ * (l ? 2 * H_ : D_));
      conv(ENC(l, d, 1), whh[l][d], G_ * H_);
    }
  conv(DEC(0, 0), dwih1, G_ * D_);
  conv(DEC(0, 1), dwhh1, G_ * H_);
  conv(DEC(1, 0), dwih2, G_ * H_);
  conv(DEC(1, 1), dwhh2, G_ * H_);
  conv(Wout, woutb, D_ * H_);
  conv(Wlat, wlatb, Z_ * 2 * H_);
  conv(Wz2h, wz2hb, 2 * H_ * Z_);

  // encoder: per layer, chip-wide gx GEMMs then batch-parallel scans
  const bf16* in_ptr = xbf;
  int K = D_;
  for (int l = 0; l < 3; ++l) {
    const int Mrows = T_ * B_;
    const int blocks = ((Mrows / 16) * (G_ / 16) + 7) / 8;
    gemm_wmma<true><<<blocks, 256, 0, stream>>>(in_ptr, K, wih[l][0],
                                                ENC(l, 0, 2), gx0, G_, Mrows, G_, K);
    gemm_wmma<true><<<blocks, 256, 0, stream>>>(in_ptr, K, wih[l][1],
                                                ENC(l, 1, 2), gx1, G_, Mrows, G_, K);
    bf16* ys_out = (l == 1) ? ysB : ysA;  // l0->ysA, l1->ysB, l2->ysA
    enc_scan<<<8, 256, 0, stream>>>(gx0, gx1, whh[l][0], whh[l][1],
                                    ENC(l, 0, 3), ENC(l, 1, 3), ys_out, comb);
    in_ptr = ys_out;
    K = 2 * H_;
  }

  // latent: z = tanh(LN(comb @ Wlat^T + blat)) ; hid = z @ Wz2h^T + bz2h
  conv(comb, combbf, B_ * 2 * H_);
  gemm_wmma<false><<<((B_ / 16) * (Z_ / 16) + 7) / 8, 256, 0, stream>>>(
      combbf, 2 * H_, wlatb, blat, zpre, Z_, B_, Z_, 2 * H_);
  float* zout = (float*)d_out + (size_t)B_ * T_ * D_;
  latent_ln<<<B_, 128, 0, stream>>>(zpre, lng, lnb, zout, zbf);
  gemm_wmma<false><<<((B_ / 16) * ((2 * H_) / 16) + 7) / 8, 256, 0, stream>>>(
      zbf, Z_, wz2hb, bz2h, hid, 2 * H_, B_, 2 * H_, Z_);

  // decoder
  dec_scan<<<4, 256, 0, stream>>>(X, hid, dwih1, dwhh1, DEC(0, 2), DEC(0, 3),
                                  dwih2, dwhh2, DEC(1, 2), DEC(1, 3), woutb,
                                  bout, (float*)d_out);
}
